// FKAConvEncoder2d_40544491274268
// MI455X (gfx1250) — matrix operand order repaired and checked by reading, not compiled, not run on hardware
//
#include <hip/hip_runtime.h>
#include <hip/hip_bf16.h>

typedef _Float16 v16h __attribute__((ext_vector_type(16)));
typedef _Float16 v8h  __attribute__((ext_vector_type(8)));
typedef float    v8f  __attribute__((ext_vector_type(8)));

#define BB   2
#define NN   8192
#define BN   (BB * NN)
#define KNB  16
#define DIMC 64
#define HH   32
#define NBLK 5
#define RESG 64

// A-fragment mapping for 16-bit 16x32 A matrix (ISA 7.12.2):
// lane<16: halves 0..7 -> K 0..7, halves 8..15 -> K 16..23
// lane>=16: halves 0..7 -> K 8..15, halves 8..15 -> K 24..31
__device__ __forceinline__ int a_slot(int row, int k) {   // lane slot (0..31)
    return row + ((k & 8) << 1);
}
__device__ __forceinline__ int a_half(int k) {            // element index (0..15)
    return (k & 7) | ((k & 16) >> 1);
}

// Build a v16h A-fragment from f16 row-major memory: halves 0..7 and 8..15 are
// contiguous 8-half runs starting at k0+hi8 and k0+16+hi8.
__device__ __forceinline__ v16h a_frag_rowmajor(const _Float16* base, int hi8) {
    v8h lo = *(const v8h*)(base + hi8);
    v8h hi = *(const v8h*)(base + 16 + hi8);
    return __builtin_shufflevector(lo, hi, 0, 1, 2, 3, 4, 5, 6, 7,
                                   8, 9, 10, 11, 12, 13, 14, 15);
}

// ---------------------------------------------------------------- stem
__global__ void stem_kernel(const float* __restrict__ xyz,
                            const float* __restrict__ w,
                            const float* __restrict__ bias,
                            float* __restrict__ c,
                            _Float16* __restrict__ ch) {
    int g = blockIdx.x * blockDim.x + threadIdx.x;
    if (g >= BN) return;
    float p0 = xyz[g * 3 + 0], p1 = xyz[g * 3 + 1], p2 = xyz[g * 3 + 2];
    for (int o = 0; o < DIMC; ++o) {
        float v = p0 * w[0 * DIMC + o] + p1 * w[1 * DIMC + o] +
                  p2 * w[2 * DIMC + o] + bias[o];
        c[g * DIMC + o] = v;
        ch[g * DIMC + o] = (_Float16)v;
    }
}

// ---------------------------------------------------------------- knn
__global__ void knn_kernel(const float* __restrict__ xyz, int* __restrict__ idx) {
    __shared__ float sx[256 * 3];
    int g = blockIdx.x * 256 + threadIdx.x;   // global query point
    int b = g / NN;
    float qx = xyz[g * 3 + 0], qy = xyz[g * 3 + 1], qz = xyz[g * 3 + 2];
    float bd[16];
    int   bi[16];
#pragma unroll
    for (int s = 0; s < 16; ++s) { bd[s] = 3.0e38f; bi[s] = 0; }
    for (int t = 0; t < NN; t += 256) {
        int j = t + threadIdx.x;
        sx[threadIdx.x * 3 + 0] = xyz[(b * NN + j) * 3 + 0];
        sx[threadIdx.x * 3 + 1] = xyz[(b * NN + j) * 3 + 1];
        sx[threadIdx.x * 3 + 2] = xyz[(b * NN + j) * 3 + 2];
        __syncthreads();
        for (int u = 0; u < 256; ++u) {
            float dx = sx[u * 3 + 0] - qx;
            float dy = sx[u * 3 + 1] - qy;
            float dz = sx[u * 3 + 2] - qz;
            float d2 = dx * dx + dy * dy + dz * dz;
            if (d2 < bd[15]) {
                bd[15] = d2; bi[15] = t + u;
#pragma unroll
                for (int s = 15; s > 0; --s) {
                    if (bd[s] < bd[s - 1]) {
                        float td = bd[s]; bd[s] = bd[s - 1]; bd[s - 1] = td;
                        int   ti = bi[s]; bi[s] = bi[s - 1]; bi[s - 1] = ti;
                    }
                }
            }
        }
        __syncthreads();
    }
#pragma unroll
    for (int s = 0; s < 16; ++s) idx[g * 16 + s] = bi[s];
}

// ---------------------------------------------------------------- weight pack (per block)
// Pre-swizzle weights into B-fragment order [frag][lane][16 halves] so each
// B operand is one aligned v16h load shared by all waves.
__global__ void pack_kernel(const float* __restrict__ w0, const float* __restrict__ w2,
                            const float* __restrict__ wcv,
                            const float* __restrict__ fc2, const float* __restrict__ fc3,
                            _Float16* __restrict__ pw0, _Float16* __restrict__ pw2,
                            _Float16* __restrict__ pwcv,
                            _Float16* __restrict__ pfc2, _Float16* __restrict__ pfc3) {
    int e = blockIdx.x * 256 + threadIdx.x;
    if (e < 2048) {                       // pw0[((kk*2+nt)*32+lane)*16+h]
        int h = e & 15, lane = (e >> 4) & 31, nt = (e >> 9) & 1, kk = (e >> 10) & 1;
        pw0[e] = (_Float16)w0[(kk * 32 + (h & 7) + ((lane & 16) >> 1) + ((h & 8) << 1)) * HH + (lane & 15) + 16 * nt];
    } else if (e < 4096) {                // pw2[(nt*32+lane)*16+h]
        int e2 = e - 2048;
        int h = e2 & 15, lane = (e2 >> 4) & 31, nt = (e2 >> 9) & 3;
        pw2[e2] = (_Float16)w2[((h & 7) + ((lane & 16) >> 1) + ((h & 8) << 1)) * DIMC + (lane & 15) + 16 * nt];
    } else if (e < 20480) {               // pwcv[((nt*16+kk)*32+lane)*16+h]
        int e2 = e - 4096;
        int h = e2 & 15, lane = (e2 >> 4) & 31, kk = (e2 >> 9) & 15, nt = (e2 >> 13) & 1;
        int o = (lane & 15) + 16 * nt;
        pwcv[e2] = (_Float16)wcv[o * 512 + kk * 32 + (h & 7) + ((lane & 16) >> 1) + ((h & 8) << 1)];
    } else if (e < 20992) {               // pfc2[lane*16+h]  (32x16 = one fragment)
        int e2 = e - 20480;
        int h = e2 & 15, lane = (e2 >> 4) & 31;
        pfc2[e2] = (_Float16)fc2[((h & 7) + ((lane & 16) >> 1) + ((h & 8) << 1)) * 16 + (lane & 15)];
    } else if (e < 21504) {               // pfc3[lane*16+h]
        int e2 = e - 20992;
        int h = e2 & 15, lane = (e2 >> 4) & 31;
        pfc3[e2] = (_Float16)fc3[((h & 7) + ((lane & 16) >> 1) + ((h & 8) << 1)) * 16 + (lane & 15)];
    }
}

// ---------------------------------------------------------------- x = relu(c @ w0 + b0)  [WMMA]
__global__ void xproj_kernel(const _Float16* __restrict__ ch,
                             const _Float16* __restrict__ pw0,
                             const float* __restrict__ b0,
                             _Float16* __restrict__ xh) {
    int lane = threadIdx.x;
    int pt0  = blockIdx.x * 16;
    int row  = pt0 + (lane & 15);
    int hi8  = (lane & 16) >> 1;
    const v16h* pw0V = (const v16h*)pw0;
    v8f acc[2] = {};
#pragma unroll
    for (int kk = 0; kk < 2; ++kk) {
        v16h a = a_frag_rowmajor(ch + row * DIMC + kk * 32, hi8);
#pragma unroll
        for (int nt = 0; nt < 2; ++nt) {
            v16h bm = pw0V[(kk * 2 + nt) * 32 + lane];
            acc[nt] = __builtin_amdgcn_wmma_f32_16x16x32_f16(
                false, a, false, bm, (short)0, acc[nt], false, false);
        }
    }
#pragma unroll
    for (int nt = 0; nt < 2; ++nt)
#pragma unroll
        for (int v = 0; v < 8; ++v) {
            int m = v + hi8;
            int o = (lane & 15) + 16 * nt;
            float val = acc[nt][v] + b0[o];
            xh[(pt0 + m) * HH + o] = (_Float16)(val > 0.f ? val : 0.f);
        }
}

// ---------------------------------------------------------------- fused FKAConv block tail
// All three inner contractions (fc2 layer, fc3 layer, neighbor-feature
// aggregation) plus the two big GEMMs run on WMMA. Fragments are assembled
// in-place in LDS; buffers are aliased across phase lifetimes.
__global__ void fka_kernel(const float* __restrict__ xyz, const int* __restrict__ idxbuf,
                           const _Float16* __restrict__ xh, float* __restrict__ c,
                           _Float16* __restrict__ ch,
                           const float* __restrict__ fc1,
                           const _Float16* __restrict__ pfc2,
                           const _Float16* __restrict__ pfc3,
                           const _Float16* __restrict__ pwcv,
                           const float* __restrict__ alpha_p, const float* __restrict__ beta_p,
                           const _Float16* __restrict__ pw2, const float* __restrict__ b2) {
    __shared__ v16h     sBufA[16 * 32];      // A1 frag [pt][slot] (phases A..C1), then F frag [kk][slot] (D..E)
    __shared__ v16h     sBufB[16 * 32];      // A2 frag (C1..C2), then m3^T frag (C2..D)
    __shared__ _Float16 sXT[16 * 32 * 16];   // neighbor features transposed [pt][c][nbr], 16KB
    __shared__ float    sDw[16 * 16];
    __shared__ float    sMp[16 * 16];
    __shared__ v16h     sYfrag[32];

    int lane = threadIdx.x;
    int pt0  = blockIdx.x * 16;
    int hi8  = (lane & 16) >> 1;
    float av = *alpha_p, bv = *beta_p;
    const v16h* pwcvV = (const v16h*)pwcv;
    const v16h* pw2V  = (const v16h*)pw2;
    const v16h* pfc2V = (const v16h*)pfc2;
    const v16h* pfc3V = (const v16h*)pfc3;
    _Float16* A1 = (_Float16*)sBufA;
    _Float16* A2 = (_Float16*)sBufB;
    v16h vz = {};
    v8h  vz8 = {};

    // Warm caches with the packed kernel weights this wave will stream.
    __builtin_prefetch(pwcv + lane * 512, 0, 1);
    __builtin_prefetch(pw2 + lane * 64, 0, 1);

    // Phase A: local coords, dw, m1 = relu(pts@fc1) written straight into
    // A-fragment layout; gather neighbor features transposed.
    for (int it = 0; it < 8; ++it) {
        int p = it * 32 + lane;
        int pt = p >> 4, nbr = p & 15;
        int g  = pt0 + pt;
        int b  = g / NN;
        int j  = idxbuf[g * 16 + nbr];
        int gj = b * NN + j;
        float px = xyz[gj * 3 + 0] - xyz[g * 3 + 0];
        float py = xyz[gj * 3 + 1] - xyz[g * 3 + 1];
        float pz = xyz[gj * 3 + 2] - xyz[g * 3 + 2];
        float dist = sqrtf(px * px + py * py + pz * pz + 1e-12f);
        sDw[pt * 16 + nbr] = 1.0f / (1.0f + __expf(av * dist - bv));
#pragma unroll
        for (int s = 0; s < 16; ++s) {
            float m = px * fc1[0 * 16 + s] + py * fc1[1 * 16 + s] + pz * fc1[2 * 16 + s];
            A1[(pt * 32 + a_slot(nbr, s)) * 16 + a_half(s)] = (_Float16)(m > 0.f ? m : 0.f);
        }
        const v16h* xr = (const v16h*)(xh + gj * HH);
        v16h x0 = xr[0], x1 = xr[1];
#pragma unroll
        for (int cc = 0; cc < 16; ++cc) {
            sXT[(pt * 32 + cc) * 16 + nbr]      = x0[cc];
            sXT[(pt * 32 + 16 + cc) * 16 + nbr] = x1[cc];
        }
    }
    __syncthreads();

    // Phase B: normalize dw (sum to K over neighbors)
    if (lane < 16) {
        float s = 0.f;
        for (int nbr = 0; nbr < 16; ++nbr) s += sDw[lane * 16 + nbr];
        float sc = (float)KNB / (s + 1e-6f);
        for (int nbr = 0; nbr < 16; ++nbr) sDw[lane * 16 + nbr] *= sc;
    }
    __syncthreads();

    // pool mp1 = max_nbr(m1*dw)  (m1 read from fragment layout)
    for (int it = 0; it < 8; ++it) {
        int q = it * 32 + lane;
        int pt = q >> 4, s = q & 15;
        float mx = 0.f;
        for (int r = 0; r < 16; ++r) {
            float v = (float)A1[(pt * 32 + a_slot(r, s)) * 16 + a_half(s)] * sDw[pt * 16 + r];
            mx = v > mx ? v : mx;
        }
        sMp[pt * 16 + s] = mx;
    }
    __syncthreads();
    // broadcast mp1 into upper K half of A1 fragments
    for (int it = 0; it < 8; ++it) {
        int q = it * 32 + lane;
        int pt = q >> 4, s = q & 15;
        _Float16 v = (_Float16)sMp[pt * 16 + s];
        int k = 16 + s;
#pragma unroll
        for (int r = 0; r < 16; ++r)
            A1[(pt * 32 + a_slot(r, k)) * 16 + a_half(k)] = v;
    }
    __syncthreads();

    // C1: m2 = relu([m1|mp1] @ fc2) — one WMMA per point tile
    {
        v16h bf = pfc2V[lane];
        for (int pt = 0; pt < 16; ++pt) {
            v16h a = sBufA[pt * 32 + lane];
            v8f acc = {};
            acc = __builtin_amdgcn_wmma_f32_16x16x32_f16(
                false, a, false, bf, (short)0, acc, false, false);
#pragma unroll
            for (int v = 0; v < 8; ++v) {
                int r = v + hi8, s = lane & 15;
                float val = acc[v];
                val = val > 0.f ? val : 0.f;
                A2[(pt * 32 + a_slot(r, s)) * 16 + a_half(s)] = (_Float16)val;
            }
        }
    }
    __syncthreads();

    // pool mp2 = max_nbr(m2*dw)
    for (int it = 0; it < 8; ++it) {
        int q = it * 32 + lane;
        int pt = q >> 4, s = q & 15;
        float mx = 0.f;
        for (int r = 0; r < 16; ++r) {
            float v = (float)A2[(pt * 32 + a_slot(r, s)) * 16 + a_half(s)] * sDw[pt * 16 + r];
            mx = v > mx ? v : mx;
        }
        sMp[pt * 16 + s] = mx;
    }
    __syncthreads();
    // broadcast mp2 into upper K half of A2 fragments
    for (int it = 0; it < 8; ++it) {
        int q = it * 32 + lane;
        int pt = q >> 4, s = q & 15;
        _Float16 v = (_Float16)sMp[pt * 16 + s];
        int k = 16 + s;
#pragma unroll
        for (int r = 0; r < 16; ++r)
            A2[(pt * 32 + a_slot(r, k)) * 16 + a_half(k)] = v;
    }
    __syncthreads();

    // C2: m3 = relu([m2|mp2] @ fc3) * dw — written back transposed (rows=s,
    // K=nbr) into the same buffer as the A fragments for phase D.
    {
        v16h bf = pfc3V[lane];
        for (int pt = 0; pt < 16; ++pt) {
            v16h a = sBufB[pt * 32 + lane];
            v8f acc = {};
            acc = __builtin_amdgcn_wmma_f32_16x16x32_f16(
                false, a, false, bf, (short)0, acc, false, false);
#pragma unroll
            for (int v = 0; v < 8; ++v) {
                int r = v + hi8, s = lane & 15;
                float val = acc[v];
                val = val > 0.f ? val : 0.f;
                val *= sDw[pt * 16 + r];
                A2[(pt * 32 + a_slot(s, r)) * 16 + a_half(r)] = (_Float16)val;
            }
        }
    }
    __syncthreads();
    // zero the (stale) upper K half of the m3^T fragments
#pragma unroll
    for (int pt = 0; pt < 16; ++pt)
        *(v8h*)((_Float16*)&sBufB[pt * 32 + lane] + 8) = vz8;
    __syncthreads();

    // Phase D: F^T = m3^T @ nb_x per point (K=16 zero-padded to 32),
    // scattered into the phase-E A-fragment layout.
    for (int pt = 0; pt < 16; ++pt) {
        v16h a = sBufB[pt * 32 + lane];
#pragma unroll
        for (int nt = 0; nt < 2; ++nt) {
            int cidx = (lane & 15) + 16 * nt;
            v8h t = *(const v8h*)(sXT + (pt * 32 + cidx) * 16 + hi8);
            v16h bm = __builtin_shufflevector(t, vz8, 0, 1, 2, 3, 4, 5, 6, 7, 8, 9, 10, 11, 12, 13, 14, 15);
            v8f acc = {};
            acc = __builtin_amdgcn_wmma_f32_16x16x32_f16(
                false, a, false, bm, (short)0, acc, false, false);
#pragma unroll
            for (int v = 0; v < 8; ++v) {
                int s  = v + hi8;
                int cs = cidx * 16 + s;
                int kk = cs >> 5, kin = cs & 31;
                ((_Float16*)&sBufA[kk * 32 + a_slot(pt, kin)])[a_half(kin)] = (_Float16)acc[v];
            }
        }
    }
    __syncthreads();

    // Phase E: y = relu(F @ Wcv2), K=512 in 16 fully-unrolled WMMA steps.
    {
        v8f acc[2] = {};
#pragma unroll
        for (int kk = 0; kk < 16; ++kk) {
            v16h a = sBufA[kk * 32 + lane];
#pragma unroll
            for (int nt = 0; nt < 2; ++nt) {
                v16h bm = pwcvV[(nt * 16 + kk) * 32 + lane];
                acc[nt] = __builtin_amdgcn_wmma_f32_16x16x32_f16(
                    false, a, false, bm, (short)0, acc[nt], false, false);
            }
        }
#pragma unroll
        for (int nt = 0; nt < 2; ++nt)
#pragma unroll
            for (int v = 0; v < 8; ++v) {
                int m = v + hi8;
                int o = (lane & 15) + 16 * nt;
                float val = acc[nt][v];
                val = val > 0.f ? val : 0.f;
                ((_Float16*)&sYfrag[a_slot(m, o)])[a_half(o)] = (_Float16)val;
            }
    }
    __syncthreads();

    // Phase F: c = relu(y @ w2 + b2 + c)   (K=32, N=64 in 4 tiles)
    {
        v16h a = sYfrag[lane];
#pragma unroll
        for (int nt = 0; nt < 4; ++nt) {
            v8f acc = {};
            v16h bm = pw2V[nt * 32 + lane];
            acc = __builtin_amdgcn_wmma_f32_16x16x32_f16(
                false, a, false, bm, (short)0, acc, false, false);
#pragma unroll
            for (int v = 0; v < 8; ++v) {
                int m = v + hi8;
                int o = (lane & 15) + 16 * nt;
                int g = pt0 + m;
                float val = acc[v] + b2[o] + c[g * DIMC + o];
                val = val > 0.f ? val : 0.f;
                c[g * DIMC + o] = val;
                ch[g * DIMC + o] = (_Float16)val;
            }
        }
    }
}

// ---------------------------------------------------------------- outputs
__global__ void copyout_kernel(const float* __restrict__ xyz,
                               const float* __restrict__ c,
                               float* __restrict__ out) {
    int e = blockIdx.x * 256 + threadIdx.x;
    if (e < BN * 3) out[e] = xyz[e];
    else if (e < BN * 3 + BN * DIMC) out[e] = c[e - BN * 3];
}

__global__ void zero_kernel(float* __restrict__ p, int n) {
    int e = blockIdx.x * 256 + threadIdx.x;
    if (e < n) p[e] = 0.f;
}

__global__ void tri_scatter(const float* __restrict__ xyz, const float* __restrict__ c,
                            float* __restrict__ sums, float* __restrict__ cnts) {
    int g = blockIdx.x * 256 + threadIdx.x;
    if (g >= BN) return;
    int b = g / NN;
    int ij[3];
    for (int d = 0; d < 3; ++d) {
        float pn = xyz[g * 3 + d] / (1.0f + 0.1f + 1e-3f) + 0.5f;
        int v = (int)(pn * (float)RESG);
        ij[d] = v < 0 ? 0 : (v > RESG - 1 ? RESG - 1 : v);
    }
    const int pa[3] = {0, 0, 1};
    const int pb[3] = {1, 2, 2};
    for (int p = 0; p < 3; ++p) {
        int cell = (b * 3 + p) * RESG * RESG + ij[pa[p]] * RESG + ij[pb[p]];
        atomicAdd(&cnts[cell], 1.0f);
        for (int d = 0; d < DIMC; ++d)
            atomicAdd(&sums[cell * DIMC + d], c[g * DIMC + d]);
    }
}

__global__ void tri_final(const float* __restrict__ sums, const float* __restrict__ cnts,
                          float* __restrict__ out) {
    int e = blockIdx.x * 256 + threadIdx.x;
    if (e >= BB * 3 * RESG * RESG * DIMC) return;
    int cell = e >> 6;
    float cnt = cnts[cell];
    out[e] = sums[e] / (cnt > 1.0f ? cnt : 1.0f);
}

// ---------------------------------------------------------------- launch
extern "C" void kernel_launch(void* const* d_in, const int* in_sizes, int n_in,
                              void* d_out, int out_size, void* d_ws, size_t ws_size,
                              hipStream_t stream) {
    const float* xyz    = (const float*)d_in[0];
    const float* w_stem = (const float*)d_in[1];
    const float* b_stem = (const float*)d_in[2];
    const float* w0     = (const float*)d_in[3];
    const float* b0     = (const float*)d_in[4];
    const float* fc1    = (const float*)d_in[5];
    const float* fc2    = (const float*)d_in[6];
    const float* fc3    = (const float*)d_in[7];
    const float* wcv    = (const float*)d_in[8];
    const float* alpha  = (const float*)d_in[9];
    const float* beta   = (const float*)d_in[10];
    const float* w2     = (const float*)d_in[11];
    const float* b2     = (const float*)d_in[12];

    float* ws = (float*)d_ws;
    float*    cbuf = ws;                                            // BN*64 f32
    float*    sums = cbuf + (size_t)BN * DIMC;                      // 2*3*64*64*64
    float*    cnts = sums + (size_t)BB * 3 * RESG * RESG * DIMC;    // 2*3*64*64
    int*      idxb = (int*)(cnts + (size_t)BB * 3 * RESG * RESG);   // BN*16 int
    _Float16* ch   = (_Float16*)(idxb + (size_t)BN * KNB);          // BN*64 f16
    _Float16* xh   = ch + (size_t)BN * DIMC;                        // BN*32 f16
    _Float16* pw0  = xh + (size_t)BN * HH;                          // 2048 f16
    _Float16* pw2  = pw0 + 2048;                                    // 2048 f16
    _Float16* pwcv = pw2 + 2048;                                    // 16384 f16
    _Float16* pfc2 = pwcv + 16384;                                  // 512 f16
    _Float16* pfc3 = pfc2 + 512;                                    // 512 f16

    stem_kernel<<<BN / 256, 256, 0, stream>>>(xyz, w_stem, b_stem, cbuf, ch);
    knn_kernel<<<BN / 256, 256, 0, stream>>>(xyz, idxb);

    for (int nb = 0; nb < NBLK; ++nb) {
        pack_kernel<<<84, 256, 0, stream>>>(
            w0 + nb * DIMC * HH, w2 + nb * HH * DIMC, wcv + nb * HH * HH * 16,
            fc2 + nb * 32 * 16, fc3 + nb * 32 * 16,
            pw0, pw2, pwcv, pfc2, pfc3);
        xproj_kernel<<<BN / 16, 32, 0, stream>>>(ch, pw0, b0 + nb * HH, xh);
        fka_kernel<<<BN / 16, 32, 0, stream>>>(
            xyz, idxb, xh, cbuf, ch,
            fc1 + nb * 3 * 16, pfc2, pfc3,
            pwcv, alpha + nb, beta + nb, pw2, b2 + nb * DIMC);
    }

    copyout_kernel<<<(BN * 3 + BN * DIMC + 255) / 256, 256, 0, stream>>>(xyz, cbuf, (float*)d_out);

    int nz = BB * 3 * RESG * RESG * DIMC + BB * 3 * RESG * RESG;
    zero_kernel<<<(nz + 255) / 256, 256, 0, stream>>>(sums, nz);
    tri_scatter<<<BN / 256, 256, 0, stream>>>(xyz, cbuf, sums, cnts);
    tri_final<<<(BB * 3 * RESG * RESG * DIMC + 255) / 256, 256, 0, stream>>>(
        sums, cnts, (float*)d_out + BN * 3 + BN * DIMC);
}
